// GGNN_43568148251382
// MI455X (gfx1250) — matrix operand order repaired and checked by reading, not compiled, not run on hardware
//
#include <hip/hip_runtime.h>
#include <hip/hip_bf16.h>

// ---------------------------------------------------------------------------
// GGNN forward for MI455X (gfx1250, wave32, WMMA 16x16x32 f16 -> f32)
// ---------------------------------------------------------------------------

typedef __attribute__((ext_vector_type(16))) _Float16 v16h;
typedef __attribute__((ext_vector_type(8)))  _Float16 v8h;
typedef __attribute__((ext_vector_type(2)))  _Float16 v2h;
typedef __attribute__((ext_vector_type(8)))  float    v8f;

#define F_IN    256
#define HDIM    64
#define CDIM    40
#define CPAD    48
#define LLAYERS 4

// ---------------------------------------------------------------------------
// WMMA helpers
// ---------------------------------------------------------------------------
__device__ __forceinline__ v8f wmma_f16(v16h a, v16h b, v8f c) {
    // emits v_wmma_f32_16x16x32_f16
    return __builtin_amdgcn_wmma_f32_16x16x32_f16(
        /*neg_a=*/false, a, /*neg_b=*/false, b,
        /*c_mod=*/(short)0, c, /*reuse_a=*/false, /*reuse_b=*/false);
}

// A fragment (16x32 f16), lane L<16: M=L, K = kb+{0..7, 16..23};
// lane L>=16: M=L-16, K = kb+{8..15, 24..31}. rowptr already points at A[m].
__device__ __forceinline__ v16h ldA16(const _Float16* rowptr, int kb, int hi) {
    const _Float16* p = rowptr + kb + hi * 8;
    v8h lo = *(const v8h*)(p);
    v8h hh = *(const v8h*)(p + 16);
    return __builtin_shufflevector(lo, hh, 0,1,2,3,4,5,6,7,8,9,10,11,12,13,14,15);
}

// Same A fragment but converting from an f32 row on the fly.
__device__ __forceinline__ v16h ldA32(const float* rowptr, int kb, int hi) {
    const float* p = rowptr + kb + hi * 8;
    v16h a;
#pragma unroll
    for (int i = 0; i < 8; ++i) {
        a[i]     = (_Float16)p[i];
        a[i + 8] = (_Float16)p[16 + i];
    }
    return a;
}

// B fragment (32x16 f16): lanes 0-15 hold K=kb+{0..7, 16..23} of column N=lane,
// lanes 16-31 hold K=kb+{8..15, 24..31}. Source is B^T stored [Ncols][K] row-major
// (ld = K in halves), so each lane reads two runs of 8 contiguous halves.
__device__ __forceinline__ v16h ldB(const _Float16* bt, int ld, int nbase, int kb, int lane) {
    int n  = lane & 15;
    int kh = lane >> 4;
    const _Float16* p = bt + (size_t)(nbase + n) * ld + kb + kh * 8;
    v8h lo = *(const v8h*)(p);
    v8h hh = *(const v8h*)(p + 16);
    return __builtin_shufflevector(lo, hh, 0,1,2,3,4,5,6,7,8,9,10,11,12,13,14,15);
}

// ---------------------------------------------------------------------------
// Weight prep: convert to f16 and (where needed) transpose into B^T layout.
// ---------------------------------------------------------------------------
__global__ void k_prep(const float* __restrict__ Win, const float* __restrict__ Wc,
                       const float* __restrict__ Wih, const float* __restrict__ Whh,
                       const float* __restrict__ Wout,
                       _Float16* __restrict__ WinT, _Float16* __restrict__ WcT,
                       _Float16* __restrict__ Wih16, _Float16* __restrict__ Whh16,
                       _Float16* __restrict__ WoutT) {
    int t = blockIdx.x * blockDim.x + threadIdx.x;
    // W_in [F,H] -> WinT [H][F]  (B^T)
    if (t < HDIM * F_IN) {
        int n = t / F_IN, k = t % F_IN;
        WinT[t] = (_Float16)Win[k * HDIM + n];
        return;
    }
    t -= HDIM * F_IN;
    // conv_weight [L,H,H] -> WcT [L][Hout][Hin]  (per-layer B^T)
    if (t < LLAYERS * HDIM * HDIM) {
        int l = t / (HDIM * HDIM), r = t % (HDIM * HDIM);
        int n = r / HDIM, k = r % HDIM;
        WcT[t] = (_Float16)Wc[l * HDIM * HDIM + k * HDIM + n];
        return;
    }
    t -= LLAYERS * HDIM * HDIM;
    // w_ih [3H,H]: reference uses a @ w_ih.T, so w_ih as stored IS B^T.
    if (t < 3 * HDIM * HDIM) { Wih16[t] = (_Float16)Wih[t]; return; }
    t -= 3 * HDIM * HDIM;
    if (t < 3 * HDIM * HDIM) { Whh16[t] = (_Float16)Whh[t]; return; }
    t -= 3 * HDIM * HDIM;
    // weight_out [H,C] -> WoutT [CPAD][H] (B^T, zero-padded rows 40..47)
    if (t < CPAD * HDIM) {
        int n = t / HDIM, k = t % HDIM;
        WoutT[t] = (n < CDIM) ? (_Float16)Wout[k * CDIM + n] : (_Float16)0.0f;
    }
}

// ---------------------------------------------------------------------------
// h = x @ W_in   (f32 A converted on the fly; writes h f32 + h16 f16 mirror)
// ---------------------------------------------------------------------------
__global__ void k_proj_in(const float* __restrict__ x, const _Float16* __restrict__ WinT,
                          float* __restrict__ h, _Float16* __restrict__ h16, int Nn) {
    int lane = threadIdx.x & 31;
    int gw   = blockIdx.x * (blockDim.x >> 5) + (threadIdx.x >> 5);
    int rowbase = gw * 16;
    if (rowbase >= Nn) return;               // uniform per wave -> EXEC stays full
    int r = rowbase + (lane & 15);
    if (r > Nn - 1) r = Nn - 1;
    int hi = lane >> 4;
    const float* arow = x + (size_t)r * F_IN;

    v8f acc[4] = {};
#pragma unroll
    for (int kb = 0; kb < F_IN; kb += 32) {
        v16h a = ldA32(arow, kb, hi);
#pragma unroll
        for (int j = 0; j < 4; ++j) {
            v16h b = ldB(WinT, F_IN, j * 16, kb, lane);
            acc[j] = wmma_f16(a, b, acc[j]);
        }
    }
    int col0 = lane & 15;
#pragma unroll
    for (int j = 0; j < 4; ++j) {
        int col = j * 16 + col0;
#pragma unroll
        for (int i = 0; i < 8; ++i) {
            int row = rowbase + i + 8 * hi;
            if (row < Nn) {
                float v = acc[j][i];
                h[(size_t)row * HDIM + col]   = v;
                h16[(size_t)row * HDIM + col] = (_Float16)v;
            }
        }
    }
}

// ---------------------------------------------------------------------------
// m = h @ conv_weight[l]  (f16 in, f16 out to halve edge-gather bytes)
// ---------------------------------------------------------------------------
__global__ void k_conv(const _Float16* __restrict__ h16, const _Float16* __restrict__ WcTl,
                       _Float16* __restrict__ m16, int Nn) {
    int lane = threadIdx.x & 31;
    int gw   = blockIdx.x * (blockDim.x >> 5) + (threadIdx.x >> 5);
    int rowbase = gw * 16;
    if (rowbase >= Nn) return;
    int r = rowbase + (lane & 15);
    if (r > Nn - 1) r = Nn - 1;
    int hi = lane >> 4;
    const _Float16* arow = h16 + (size_t)r * HDIM;

    v16h a0 = ldA16(arow, 0, hi);
    v16h a1 = ldA16(arow, 32, hi);
    v8f acc[4] = {};
#pragma unroll
    for (int j = 0; j < 4; ++j) {
        acc[j] = wmma_f16(a0, ldB(WcTl, HDIM, j * 16, 0, lane), acc[j]);
        acc[j] = wmma_f16(a1, ldB(WcTl, HDIM, j * 16, 32, lane), acc[j]);
    }
    int col0 = lane & 15;
#pragma unroll
    for (int j = 0; j < 4; ++j) {
        int col = j * 16 + col0;
#pragma unroll
        for (int i = 0; i < 8; ++i) {
            int row = rowbase + i + 8 * hi;
            if (row < Nn) m16[(size_t)row * HDIM + col] = (_Float16)acc[j][i];
        }
    }
}

// ---------------------------------------------------------------------------
// agg zero + edge scatter-add (f16 gather, hardware f32 atomic accumulate;
// agg (25.6 MB) is L2-resident, so the atomics never leave the 192MB L2).
// ---------------------------------------------------------------------------
__global__ void k_zero4(float4* __restrict__ p, int n4) {
    int t = blockIdx.x * blockDim.x + threadIdx.x;
    if (t < n4) p[t] = make_float4(0.f, 0.f, 0.f, 0.f);
}

__global__ void k_scatter(const _Float16* __restrict__ m16, const int* __restrict__ ei,
                          float* __restrict__ agg, int Ee) {
    long long t = (long long)blockIdx.x * blockDim.x + threadIdx.x;
    int e = (int)(t >> 5);
    if (e >= Ee) return;
    int lane = (int)(t & 31);
    int s = ei[e];
    int d = ei[Ee + e];
    const _Float16* mp = m16 + (size_t)s * HDIM + lane * 2;
    __builtin_prefetch(mp, 0, 1);            // global_prefetch_b8: overlap index latency
    v2h mv = *(const v2h*)mp;
    float* ap = agg + (size_t)d * HDIM + lane * 2;
    // unsafeAtomicAdd guarantees global_atomic_add_f32 (no CAS loop) on the
    // bottleneck kernel: 2 fire-and-forget f32 atomics per lane per edge.
    unsafeAtomicAdd(ap,     (float)mv[0]);
    unsafeAtomicAdd(ap + 1, (float)mv[1]);
}

// ---------------------------------------------------------------------------
// Fused GRU cell: gi = agg@W_ih^T + b_ih ; gh = h@W_hh^T + b_hh ; gate math.
// 48 WMMA per wave; accumulator layout element i <-> (M = i+8*hi, N = lane&15)
// ---------------------------------------------------------------------------
__global__ void k_gru(const float* __restrict__ agg, float* __restrict__ h,
                      _Float16* __restrict__ h16,
                      const _Float16* __restrict__ Wih16, const _Float16* __restrict__ Whh16,
                      const float* __restrict__ bih, const float* __restrict__ bhh, int Nn) {
    int lane = threadIdx.x & 31;
    int gw   = blockIdx.x * (blockDim.x >> 5) + (threadIdx.x >> 5);
    int rowbase = gw * 16;
    if (rowbase >= Nn) return;
    int r = rowbase + (lane & 15);
    if (r > Nn - 1) r = Nn - 1;
    int hi = lane >> 4;

    const float*    arow = agg + (size_t)r * HDIM;
    const _Float16* hrow = h16 + (size_t)r * HDIM;
    v16h aA0 = ldA32(arow, 0, hi), aA1 = ldA32(arow, 32, hi);
    v16h hA0 = ldA16(hrow, 0, hi), hA1 = ldA16(hrow, 32, hi);

    int col0 = lane & 15;
#pragma unroll
    for (int j = 0; j < 4; ++j) {
        v8f ir = {}, iz = {}, in_ = {}, hr = {}, hz = {}, hn = {};
        ir  = wmma_f16(aA0, ldB(Wih16, HDIM,       j * 16, 0, lane), ir);
        iz  = wmma_f16(aA0, ldB(Wih16, HDIM,  64 + j * 16, 0, lane), iz);
        in_ = wmma_f16(aA0, ldB(Wih16, HDIM, 128 + j * 16, 0, lane), in_);
        hr  = wmma_f16(hA0, ldB(Whh16, HDIM,       j * 16, 0, lane), hr);
        hz  = wmma_f16(hA0, ldB(Whh16, HDIM,  64 + j * 16, 0, lane), hz);
        hn  = wmma_f16(hA0, ldB(Whh16, HDIM, 128 + j * 16, 0, lane), hn);
        ir  = wmma_f16(aA1, ldB(Wih16, HDIM,       j * 16, 32, lane), ir);
        iz  = wmma_f16(aA1, ldB(Wih16, HDIM,  64 + j * 16, 32, lane), iz);
        in_ = wmma_f16(aA1, ldB(Wih16, HDIM, 128 + j * 16, 32, lane), in_);
        hr  = wmma_f16(hA1, ldB(Whh16, HDIM,       j * 16, 32, lane), hr);
        hz  = wmma_f16(hA1, ldB(Whh16, HDIM,  64 + j * 16, 32, lane), hz);
        hn  = wmma_f16(hA1, ldB(Whh16, HDIM, 128 + j * 16, 32, lane), hn);

        int col = j * 16 + col0;
        float bir = bih[col], biz = bih[64 + col], bin = bih[128 + col];
        float bhr = bhh[col], bhz = bhh[64 + col], bhn = bhh[128 + col];
#pragma unroll
        for (int i = 0; i < 8; ++i) {
            int row = rowbase + i + 8 * hi;
            if (row < Nn) {
                size_t idx = (size_t)row * HDIM + col;
                float hold = h[idx];
                float rg = 1.f / (1.f + expf(-(ir[i] + bir + hr[i] + bhr)));
                float zg = 1.f / (1.f + expf(-(iz[i] + biz + hz[i] + bhz)));
                float ng = tanhf(in_[i] + bin + rg * (hn[i] + bhn));
                float hv = (1.f - zg) * ng + zg * hold;
                h[idx]   = hv;
                h16[idx] = (_Float16)hv;
            }
        }
    }
}

// ---------------------------------------------------------------------------
// out = log_softmax(h @ W_out): 64x48 padded WMMA + shfl_xor row reductions
// within each 16-lane half (rows live in fixed lane halves on wave32).
// ---------------------------------------------------------------------------
__global__ void k_out(const _Float16* __restrict__ h16, const _Float16* __restrict__ WoutT,
                      float* __restrict__ out, int Nn) {
    int lane = threadIdx.x & 31;
    int gw   = blockIdx.x * (blockDim.x >> 5) + (threadIdx.x >> 5);
    int rowbase = gw * 16;
    if (rowbase >= Nn) return;
    int r = rowbase + (lane & 15);
    if (r > Nn - 1) r = Nn - 1;
    int hi = lane >> 4;
    const _Float16* arow = h16 + (size_t)r * HDIM;

    v16h a0 = ldA16(arow, 0, hi), a1 = ldA16(arow, 32, hi);
    v8f acc[3] = {};
#pragma unroll
    for (int j = 0; j < 3; ++j) {
        acc[j] = wmma_f16(a0, ldB(WoutT, HDIM, j * 16, 0, lane), acc[j]);
        acc[j] = wmma_f16(a1, ldB(WoutT, HDIM, j * 16, 32, lane), acc[j]);
    }
    int col0 = lane & 15;
    bool v2ok = (32 + col0) < CDIM;
#pragma unroll
    for (int i = 0; i < 8; ++i) {
        float v0 = acc[0][i], v1 = acc[1][i], v2 = acc[2][i];
        float m = fmaxf(v0, v1);
        if (v2ok) m = fmaxf(m, v2);
#pragma unroll
        for (int off = 1; off < 16; off <<= 1) m = fmaxf(m, __shfl_xor(m, off, 32));
        float s = expf(v0 - m) + expf(v1 - m) + (v2ok ? expf(v2 - m) : 0.f);
#pragma unroll
        for (int off = 1; off < 16; off <<= 1) s += __shfl_xor(s, off, 32);
        float lse = m + logf(s);
        int row = rowbase + i + 8 * hi;
        if (row < Nn) {
            out[(size_t)row * CDIM + col0]      = v0 - lse;
            out[(size_t)row * CDIM + 16 + col0] = v1 - lse;
            if (v2ok) out[(size_t)row * CDIM + 32 + col0] = v2 - lse;
        }
    }
}

// ---------------------------------------------------------------------------
// Launch
// ---------------------------------------------------------------------------
extern "C" void kernel_launch(void* const* d_in, const int* in_sizes, int n_in,
                              void* d_out, int out_size, void* d_ws, size_t ws_size,
                              hipStream_t stream) {
    const float* x    = (const float*)d_in[0];
    const int*   ei   = (const int*)d_in[1];
    const float* Win  = (const float*)d_in[2];
    const float* Wout = (const float*)d_in[3];
    const float* Wc   = (const float*)d_in[4];
    const float* Wih  = (const float*)d_in[5];
    const float* Whh  = (const float*)d_in[6];
    const float* bih  = (const float*)d_in[7];
    const float* bhh  = (const float*)d_in[8];
    float* out = (float*)d_out;

    int Nn = in_sizes[0] / F_IN;   // 100000
    int Ee = in_sizes[1] / 2;      // 1200000
    (void)n_in; (void)out_size; (void)ws_size;

    // carve workspace (256B aligned)
    char* ws = (char*)d_ws;
    size_t off = 0;
    auto carve = [&](size_t bytes) -> void* {
        void* p = ws + off;
        off = (off + bytes + 255) & ~(size_t)255;
        return p;
    };
    _Float16* WinT  = (_Float16*)carve((size_t)HDIM * F_IN * 2);
    _Float16* WcT   = (_Float16*)carve((size_t)LLAYERS * HDIM * HDIM * 2);
    _Float16* Wih16 = (_Float16*)carve((size_t)3 * HDIM * HDIM * 2);
    _Float16* Whh16 = (_Float16*)carve((size_t)3 * HDIM * HDIM * 2);
    _Float16* WoutT = (_Float16*)carve((size_t)CPAD * HDIM * 2);
    float*    h     = (float*)carve((size_t)Nn * HDIM * 4);
    _Float16* h16   = (_Float16*)carve((size_t)Nn * HDIM * 2);
    _Float16* m16   = (_Float16*)carve((size_t)Nn * HDIM * 2);
    float*    agg   = (float*)carve((size_t)Nn * HDIM * 4);

    int prepTot = HDIM * F_IN + LLAYERS * HDIM * HDIM + 6 * HDIM * HDIM + CPAD * HDIM;
    k_prep<<<(prepTot + 255) / 256, 256, 0, stream>>>(Win, Wc, Wih, Whh, Wout,
                                                      WinT, WcT, Wih16, Whh16, WoutT);

    int ngroups    = (Nn + 15) / 16;          // 16-row blocks, one wave each
    int gemmBlocks = (ngroups + 7) / 8;       // 8 waves / 256-thread block
    k_proj_in<<<gemmBlocks, 256, 0, stream>>>(x, WinT, h, h16, Nn);

    int n4 = (Nn * HDIM) / 4;
    long long sthreads = (long long)Ee * 32;
    unsigned scatBlocks = (unsigned)((sthreads + 255) / 256);

    for (int l = 0; l < LLAYERS; ++l) {
        k_conv<<<gemmBlocks, 256, 0, stream>>>(h16, WcT + (size_t)l * HDIM * HDIM, m16, Nn);
        k_zero4<<<(n4 + 255) / 256, 256, 0, stream>>>((float4*)agg, n4);
        k_scatter<<<scatBlocks, 256, 0, stream>>>(m16, ei, agg, Ee);
        k_gru<<<gemmBlocks, 256, 0, stream>>>(agg, h, h16, Wih16, Whh16, bih, bhh, Nn);
    }
    k_out<<<gemmBlocks, 256, 0, stream>>>(h16, WoutT, out, Nn);
}
